// Rhn_1846835937484
// MI455X (gfx1250) — matrix operand order repaired and checked by reading, not compiled, not run on hardware
//
#include <hip/hip_runtime.h>
#include <hip/hip_bf16.h>

// ---------------------------------------------------------------------------
// Recurrent Highway Network, MI455X (gfx1250, wave32, WMMA bf16 16x16x32)
//   B=64 T=256 D=1024 H=1024 L=4
// ---------------------------------------------------------------------------

typedef __attribute__((ext_vector_type(16))) __bf16    v16bf;
typedef __attribute__((ext_vector_type(8)))  float     v8f;
typedef __attribute__((ext_vector_type(4)))  unsigned  v4u;

union V16U { v16bf v; v4u u[2]; };

#define B_   64
#define T_   256
#define D_   1024
#define H_   1024
#define L_   4

#define NBLK 32        // one 32-column tile of H per block
#define NTHR 192       // 6 waves: 3 gates x 2 n-subtiles(16)

#define ACTS_STRIDE 33 // pad to dodge LDS bank conflicts

// dynamic LDS layout (bytes)
#define SA_OFF   0                         // bf16 [64][1024] state     (128 KB)
#define XA_OFF   (131072)                  // bf16 [64][1024] x_t       (128 KB)
#define ACT_OFF  (262144)                  // f32  [3][64][33]          (25344 B)
#define SF_OFF   (262144 + 25344)          // f32  [64][32] local state (8 KB)
#define SMEM_BYTES (262144 + 25344 + 8192) // 295680 <= 320 KB

// ---------------------------------------------------------------------------
// fp32 [B][T][D] -> bf16 [T][B][D]
__global__ void k_cvt_x(const float* __restrict__ x, __bf16* __restrict__ xo) {
  long o = (long)blockIdx.x * blockDim.x + threadIdx.x;  // output-linear
  int d = (int)(o % D_);
  long tb = o / D_;
  int b = (int)(tb % B_);
  int t = (int)(tb / B_);
  xo[o] = (__bf16)x[((long)b * T_ + t) * D_ + d];
}

// fp32 [K][N] row-major -> bf16 [N][K] row-major (transpose + convert)
__global__ void k_transpose_cvt(const float* __restrict__ src,
                                __bf16* __restrict__ dst, int K, int N) {
  __shared__ float tile[32][33];
  int n0 = blockIdx.x * 32, k0 = blockIdx.y * 32;
  int tx = threadIdx.x, ty = threadIdx.y;  // block (32,8)
  for (int i = 0; i < 32; i += 8)
    tile[ty + i][tx] = src[(long)(k0 + ty + i) * N + n0 + tx];
  __syncthreads();
  for (int i = 0; i < 32; i += 8)
    dst[(long)(n0 + ty + i) * K + k0 + tx] = (__bf16)tile[tx][ty + i];
}

// s0 -> bf16 ping buffer; zero the barrier words (ws is poisoned 0xAA)
__global__ void k_init(const float* __restrict__ s0, __bf16* __restrict__ sb,
                       unsigned* __restrict__ bar) {
  int i = blockIdx.x * blockDim.x + threadIdx.x;
  if (i < B_ * H_) sb[i] = (__bf16)s0[i];
  if (i < 2) bar[i] = 0u;
}

// ---------------------------------------------------------------------------
__device__ __forceinline__ void grid_barrier(unsigned* cnt, unsigned* gen,
                                             unsigned& lgen) {
  __threadfence();            // agent-scope release of this wave's stores
  __syncthreads();
  unsigned target = ++lgen;
  if (threadIdx.x == 0) {
    unsigned arrived = atomicAdd(cnt, 1u);
    if (arrived == NBLK - 1u) {
      atomicExch(cnt, 0u);
      atomicExch(gen, target);
    } else {
      while (atomicAdd(gen, 0u) < target) __builtin_amdgcn_s_sleep(2);
    }
  }
  __syncthreads();
  __threadfence();            // acquire side
}

__global__ void __launch_bounds__(NTHR)
k_rhn_scan(const __bf16* __restrict__ xT,   // [T][B][D] bf16
           const __bf16* __restrict__ wT,   // [3][H=N][D=K] bf16 (transposed)
           const __bf16* __restrict__ rT,   // [3][L][H=N][H=K] bf16 (transposed)
           const float* __restrict__ bh, const float* __restrict__ bt,
           const float* __restrict__ bc,    // [L][H]
           __bf16* __restrict__ sbuf,       // [2][B][H] ping-pong
           const float* __restrict__ s0,    // [B][H]
           float* __restrict__ out,         // [B][T][H] ++ [B][H]
           unsigned* __restrict__ bar) {
  extern __shared__ char smem[];
  __bf16* sA  = (__bf16*)(smem + SA_OFF);
  __bf16* xA  = (__bf16*)(smem + XA_OFF);
  float* acts = (float*)(smem + ACT_OFF);
  float* sF   = (float*)(smem + SF_OFF);

  const int tid   = threadIdx.x;
  const int wave  = tid >> 5;
  const int lane  = tid & 31;
  const int gate  = wave >> 1;           // 0=h 1=t 2=c
  const int ncolL = (wave & 1) * 16;     // local col base within block's 32
  const int ncolG = blockIdx.x * 32 + ncolL;
  const int r     = lane & 15;
  const int half  = lane >> 4;

  const float* bias = (gate == 0) ? bh : ((gate == 1) ? bt : bc);

  // init fp32 local state columns from s0
  for (int i = tid; i < B_ * 32; i += NTHR) {
    int m = i >> 5, c = i & 31;
    sF[i] = s0[m * H_ + blockIdx.x * 32 + c];
  }

  unsigned lgen = 0;
  int cur = 0;

  for (int t = 0; t < T_; ++t) {
    const __bf16* xt = xT + (long)t * (B_ * D_);
    for (int l = 0; l < L_; ++l) {
      // ---- stage state (and x_t at l==0) into LDS -------------------------
      {
        const v4u* src = (const v4u*)(sbuf + (long)cur * (B_ * H_));
        v4u* dst = (v4u*)sA;
        for (int i = tid; i < (B_ * H_) / 8; i += NTHR) dst[i] = src[i];
        if (l == 0) {
          const v4u* s2 = (const v4u*)xt;
          v4u* d2 = (v4u*)xA;
          for (int i = tid; i < (B_ * D_) / 8; i += NTHR) d2[i] = s2[i];
        }
      }
      __syncthreads();

      // ---- GEMM: pre = s @ r[g][l] (+ x_t @ w[g] at l==0) -----------------
      v8f acc[4] = {};
      {
        const __bf16* bcol =
            rT + (((long)gate * L_ + l) * H_ + (ncolG + r)) * H_;
        for (int k = 0; k < H_; k += 32) {
          const __bf16* bp = bcol + k + half * 8;
          V16U bm;
          bm.u[0] = *(const v4u*)bp;
          bm.u[1] = *(const v4u*)(bp + 16);
          __builtin_prefetch((const void*)(bp + 512), 0, 1);
#pragma unroll
          for (int mt = 0; mt < 4; ++mt) {
            const __bf16* ap = sA + (mt * 16 + r) * H_ + k + half * 8;
            V16U am;
            am.u[0] = *(const v4u*)ap;
            am.u[1] = *(const v4u*)(ap + 16);
            acc[mt] = __builtin_amdgcn_wmma_f32_16x16x32_bf16(
                false, am.v, false, bm.v, (short)0, acc[mt], false, false);
          }
        }
      }
      if (l == 0) {
        const __bf16* bcol = wT + ((long)gate * H_ + ncolG + r) * D_;
        for (int k = 0; k < D_; k += 32) {
          const __bf16* bp = bcol + k + half * 8;
          V16U bm;
          bm.u[0] = *(const v4u*)bp;
          bm.u[1] = *(const v4u*)(bp + 16);
          __builtin_prefetch((const void*)(bp + 512), 0, 1);
#pragma unroll
          for (int mt = 0; mt < 4; ++mt) {
            const __bf16* ap = xA + (mt * 16 + r) * D_ + k + half * 8;
            V16U am;
            am.u[0] = *(const v4u*)ap;
            am.u[1] = *(const v4u*)(ap + 16);
            acc[mt] = __builtin_amdgcn_wmma_f32_16x16x32_bf16(
                false, am.v, false, bm.v, (short)0, acc[mt], false, false);
          }
        }
      }

      // ---- bias + nonlinearity -> acts LDS --------------------------------
      {
        float bval = bias[l * H_ + ncolG + r];
#pragma unroll
        for (int mt = 0; mt < 4; ++mt) {
#pragma unroll
          for (int vi = 0; vi < 8; ++vi) {
            float p = acc[mt][vi] + bval;
            float a;
            if (gate == 0) {  // tanh
              float e = __expf(-2.f * p);
              a = (1.f - e) / (1.f + e);
            } else {          // sigmoid
              a = 1.f / (1.f + __expf(-p));
            }
            int m = mt * 16 + half * 8 + vi;
            acts[(gate * 64 + m) * ACTS_STRIDE + (ncolL + r)] = a;
          }
        }
      }
      __syncthreads();

      // ---- highway combine: s = h*t + s*c ---------------------------------
      {
        __bf16* snext = sbuf + (long)(cur ^ 1) * (B_ * H_);
        for (int i = tid; i < B_ * 32; i += NTHR) {
          int m = i >> 5, c = i & 31;
          float hh = acts[(0 * 64 + m) * ACTS_STRIDE + c];
          float tt = acts[(1 * 64 + m) * ACTS_STRIDE + c];
          float cc = acts[(2 * 64 + m) * ACTS_STRIDE + c];
          float sn = hh * tt + sF[i] * cc;
          sF[i] = sn;
          int col = blockIdx.x * 32 + c;
          snext[m * H_ + col] = (__bf16)sn;
          if (l == L_ - 1) out[((long)m * T_ + t) * H_ + col] = sn;
        }
      }
      __syncthreads();
      grid_barrier(&bar[0], &bar[1], lgen);
      cur ^= 1;
    }
  }

  // ---- s_final -------------------------------------------------------------
  float* sfin = out + (long)B_ * T_ * H_;
  for (int i = tid; i < B_ * 32; i += NTHR) {
    int m = i >> 5, c = i & 31;
    sfin[m * H_ + blockIdx.x * 32 + c] = sF[i];
  }
}

// ---------------------------------------------------------------------------
extern "C" void kernel_launch(void* const* d_in, const int* in_sizes, int n_in,
                              void* d_out, int out_size, void* d_ws,
                              size_t ws_size, hipStream_t stream) {
  (void)in_sizes; (void)n_in; (void)out_size; (void)ws_size;
  const float* x    = (const float*)d_in[0];
  const float* s0   = (const float*)d_in[1];
  const float* w_h  = (const float*)d_in[2];
  const float* w_t  = (const float*)d_in[3];
  const float* w_c  = (const float*)d_in[4];
  const float* rh_w = (const float*)d_in[5];
  const float* rh_b = (const float*)d_in[6];
  const float* rt_w = (const float*)d_in[7];
  const float* rt_b = (const float*)d_in[8];
  const float* rc_w = (const float*)d_in[9];
  const float* rc_b = (const float*)d_in[10];
  float* out = (float*)d_out;

  // workspace carve-up (bytes)
  char* ws = (char*)d_ws;
  __bf16* xT   = (__bf16*)(ws);                              // 33,554,432
  __bf16* wT   = (__bf16*)(ws + 33554432);                   //  6,291,456
  __bf16* rT   = (__bf16*)(ws + 39845888);                   // 25,165,824
  __bf16* sbuf = (__bf16*)(ws + 65011712);                   //    262,144
  unsigned* bar = (unsigned*)(ws + 65273856);

  // 1) x: [B][T][D] f32 -> [T][B][D] bf16
  {
    long n = (long)B_ * T_ * D_;
    k_cvt_x<<<dim3((unsigned)(n / 256)), dim3(256), 0, stream>>>(x, xT);
  }
  // 2) input weights [D][H] -> bf16 [H][D]
  {
    dim3 tb(32, 8), tg(H_ / 32, D_ / 32);
    k_transpose_cvt<<<tg, tb, 0, stream>>>(w_h, wT + 0L * H_ * D_, D_, H_);
    k_transpose_cvt<<<tg, tb, 0, stream>>>(w_t, wT + 1L * H_ * D_, D_, H_);
    k_transpose_cvt<<<tg, tb, 0, stream>>>(w_c, wT + 2L * H_ * D_, D_, H_);
  }
  // 3) recurrent weights [L][H][H] -> bf16 [3][L][H][H] transposed
  {
    const float* rw[3] = {rh_w, rt_w, rc_w};
    dim3 tb(32, 8), tg(H_ / 32, H_ / 32);
    for (int g = 0; g < 3; ++g)
      for (int l = 0; l < L_; ++l)
        k_transpose_cvt<<<tg, tb, 0, stream>>>(
            rw[g] + (long)l * H_ * H_, rT + ((long)g * L_ + l) * H_ * H_, H_,
            H_);
  }
  // 4) init state ping buffer + barrier
  k_init<<<dim3((B_ * H_) / 256), dim3(256), 0, stream>>>(s0, sbuf, bar);

  // 5) persistent scan kernel
  k_rhn_scan<<<dim3(NBLK), dim3(NTHR), SMEM_BYTES, stream>>>(
      xT, wT, rT, rh_b, rt_b, rc_b, sbuf, s0, out, bar);
}
